// MultiHeadSelfAttention_29076928593947
// MI455X (gfx1250) — matrix operand order repaired and checked by reading, not compiled, not run on hardware
//
#include <hip/hip_runtime.h>
#include <hip/hip_bf16.h>
#include <math.h>

typedef __attribute__((ext_vector_type(16))) _Float16 v16h;
typedef __attribute__((ext_vector_type(8)))  float    v8f;

#define KH    16
#define B_SZ  4
#define M_SZ  1024
#define H_SZ  1024
#define D_SZ  64

static __device__ __forceinline__ v8f wmma16(v16h a, v16h b, v8f c) {
  // D = A(16x32 f16) * B(32x16 f16) + C(16x16 f32)
  return __builtin_amdgcn_wmma_f32_16x16x32_f16(
      false, a, false, b, (short)0, c, false, false);
}

union frag_cast { v16h h; uint4 u[2]; };

// A-fragment: 16x32 f16 tile, row-major leading dim lda (16B aligned rows).
// lane(0..31): row=lane&15, halves 0..15 -> K=kb..kb+15, kb=(lane>>4)*16
static __device__ __forceinline__ v16h load_a_frag(const _Float16* A, int lda) {
  const int lane = threadIdx.x & 31;
  const int row  = lane & 15;
  const int kb   = (lane >> 4) * 16;
  const _Float16* p = A + (size_t)row * lda + kb;
  frag_cast t;
  t.u[0] = *(const uint4*)(p);
  t.u[1] = *(const uint4*)(p + 8);
  return t.h;
}

// B-fragment from pre-packed layout: 512 halves per 32x16 tile,
// lane's 16 halves stored contiguously at P + lane*16.
static __device__ __forceinline__ v16h load_b_packed(const _Float16* P) {
  const int lane = threadIdx.x & 31;
  const uint4* p = (const uint4*)(P + (size_t)lane * 16);
  frag_cast t;
  t.u[0] = p[0];
  t.u[1] = p[1];
  return t.h;
}

// B-fragment from transposed source: logical B[k][n] = T[n*ld + k]
// (K^T from row-major K; V from transposed V). T points at (n0, k0).
static __device__ __forceinline__ v16h load_bT_frag(const _Float16* T, int ld) {
  const int lane = threadIdx.x & 31;
  const int n    = lane & 15;
  const int kb   = (lane >> 4) * 16;
  const _Float16* p = T + (size_t)n * ld + kb;
  frag_cast t;
  t.u[0] = *(const uint4*)(p);
  t.u[1] = *(const uint4*)(p + 8);
  return t.h;
}

__global__ void cvt_f32_f16(const float* __restrict__ x,
                            _Float16* __restrict__ y, int n) {
  int i = blockIdx.x * blockDim.x + threadIdx.x;
  if (i < n) y[i] = (_Float16)x[i];
}

// Pack f32 matrix src[Kd x N] into WMMA-B tile layout (f16):
// dst[((kt*ntiles + nt)*32 + lane)*16 + j] = src[kt*32 + (lane>>4)*16 + j][nt*16 + (lane&15)]
__global__ void pack_b_f16(const float* __restrict__ src,
                           _Float16* __restrict__ dst, int Kd, int N) {
  int idx  = blockIdx.x * blockDim.x + threadIdx.x;
  int lane = idx & 31;
  int tile = idx >> 5;
  int ntiles = N >> 4;
  int ktiles = Kd >> 5;
  if (tile >= ktiles * ntiles) return;
  int kt = tile / ntiles, nt = tile % ntiles;
  int col  = nt * 16 + (lane & 15);
  int krow = kt * 32 + (lane >> 4) * 16;
  _Float16* d = dst + (size_t)idx * 16;
#pragma unroll
  for (int j = 0; j < 16; ++j)
    d[j] = (_Float16)src[(size_t)(krow + j) * N + col];
}

// C[Mr x N] = A[Mr x Kd](f16, row-major) * Bp(packed f16), f32 accumulate.
// One wave computes a 32 x (16*NT) tile: 2 A-frags x NT B-frags per k-step.
// No conditionals near WMMA -> EXEC stays all-1s.
template <int NT>
__global__ __launch_bounds__(256) void gemm_f16_wmma(
    const _Float16* __restrict__ A, const _Float16* __restrict__ Bp,
    float* __restrict__ Cf, _Float16* __restrict__ Ch,
    _Float16* __restrict__ ChT, int Mr, int N, int Kd) {
  const int wave = (blockIdx.x * blockDim.x + threadIdx.x) >> 5;
  const int lane = threadIdx.x & 31;
  const int nstrips = N / (16 * NT);
  const int rt = wave / nstrips;          // 32-row tile index
  const int cs = wave % nstrips;
  const int ntiles = N >> 4;

  v8f acc[2][NT];
#pragma unroll
  for (int r = 0; r < 2; ++r)
#pragma unroll
    for (int t = 0; t < NT; ++t) acc[r][t] = (v8f){};

  for (int kc = 0; kc < Kd; kc += 32) {
    const int kt = kc >> 5;
    const _Float16* Ap = A + (size_t)rt * 32 * Kd + kc;
    v16h a0 = load_a_frag(Ap, Kd);
    v16h a1 = load_a_frag(Ap + (size_t)16 * Kd, Kd);
    __builtin_prefetch(Ap + 32, 0, 1);                    // global_prefetch_b8
#pragma unroll
    for (int t = 0; t < NT; ++t) {
      v16h b = load_b_packed(Bp + (size_t)(kt * ntiles + cs * NT + t) * 512);
      acc[0][t] = wmma16(a0, b, acc[0][t]);
      acc[1][t] = wmma16(a1, b, acc[1][t]);
    }
  }

  const int hi = lane >> 4, cl = lane & 15;
#pragma unroll
  for (int rr = 0; rr < 2; ++rr)
#pragma unroll
    for (int t = 0; t < NT; ++t)
#pragma unroll
      for (int j = 0; j < 8; ++j) {
        int r = rt * 32 + rr * 16 + hi * 8 + j;
        int c = cs * (16 * NT) + t * 16 + cl;
        float v = acc[rr][t][j];
        if (Cf) Cf[(size_t)r * N + c] = v;
        if (Ch) Ch[(size_t)r * N + c] = (_Float16)v;
        // transposed store: rows are (B, M) flattened; out layout (B, H, M)
        if (ChT)
          ChT[(size_t)(r >> 10) * ((size_t)H_SZ * M_SZ) +
              (size_t)c * M_SZ + (r & (M_SZ - 1))] = (_Float16)v;
      }
}

// Fused span-gated attention, flash-style. One wave32 per (b, head, 16 rows).
// Q/K layout (B, M, H) f16 (head kh at cols kh*64..+63); Vt layout (B, H, M).
__global__ __launch_bounds__(256) void attn_fused(
    const _Float16* __restrict__ Q, const _Float16* __restrict__ Kmat,
    const _Float16* __restrict__ Vt, const float* __restrict__ Span,
    const _Float16* __restrict__ PeP, _Float16* __restrict__ O) {
  __shared__ _Float16 lds[8 * 16 * 32];      // 1 KiB per wave: P re-layout
  const int widx = threadIdx.x >> 5;
  const int lane = threadIdx.x & 31;
  const int wave = blockIdx.x * 8 + widx;
  const int mt   = wave & 63;                // M/16 row tiles
  const int head = wave >> 6;                // 0..63
  const int b    = head >> 4;
  const int kh   = head & 15;
  const int m0   = mt * 16;

  const _Float16* Qh  = Q    + (size_t)b * M_SZ * H_SZ + (size_t)kh * D_SZ;
  const _Float16* Kh  = Kmat + (size_t)b * M_SZ * H_SZ + (size_t)kh * D_SZ;
  const _Float16* Vth = Vt   + (size_t)b * H_SZ * M_SZ + (size_t)kh * D_SZ * M_SZ;
  _Float16* plds = lds + widx * (16 * 32);

  v16h qa0 = load_a_frag(Qh + (size_t)m0 * H_SZ, H_SZ);
  v16h qa1 = load_a_frag(Qh + (size_t)m0 * H_SZ + 32, H_SZ);

  const int hi = lane >> 4, cl = lane & 15;

  float mean_[8], icep[8];
#pragma unroll
  for (int j = 0; j < 8; ++j) {
    int m = m0 + hi * 8 + j;
    const float* sp = Span + ((size_t)b * M_SZ + m) * (2 * KH) + 2 * kh;
    mean_[j] = sp[0];
    icep[j]  = sp[1];
  }

  float rmax[8], rsum[8];
  v8f o[4];
#pragma unroll
  for (int t = 0; t < 4; ++t) o[t] = (v8f){};
#pragma unroll
  for (int j = 0; j < 8; ++j) { rmax[j] = -1e30f; rsum[j] = 0.f; }
  const float inv_sqrt_d = 0.125f;

  for (int n0 = 0; n0 < M_SZ; n0 += 32) {
    // ---- scores: two 16x16 subtiles; content (Q.K^T) + positional (Q.Pe) ----
    v8f s[2];
#pragma unroll
    for (int sub = 0; sub < 2; ++sub) {
      int nn = n0 + sub * 16;
      s[sub] = (v8f){};
      s[sub] = wmma16(qa0, load_bT_frag(Kh + (size_t)nn * H_SZ,      H_SZ), s[sub]);
      s[sub] = wmma16(qa1, load_bT_frag(Kh + (size_t)nn * H_SZ + 32, H_SZ), s[sub]);
      s[sub] = wmma16(qa0, load_b_packed(PeP + (size_t)(0 * 64 + (nn >> 4)) * 512), s[sub]);
      s[sub] = wmma16(qa1, load_b_packed(PeP + (size_t)(1 * 64 + (nn >> 4)) * 512), s[sub]);
    }
    // ---- span gate + scale ----
    float sv[2][8];
#pragma unroll
    for (int sub = 0; sub < 2; ++sub)
#pragma unroll
      for (int j = 0; j < 8; ++j) {
        int m = m0 + hi * 8 + j;
        int n = n0 + sub * 16 + cl;
        float t = ((float)(n - m) + mean_[j]) * 0.1f;
        float y = fminf(fmaxf(-t * t + icep[j], 0.f), 1.f);
        sv[sub][j] = s[sub][j] * y * inv_sqrt_d;
      }
    // ---- online softmax (row reductions across 16-lane halves) ----
#pragma unroll
    for (int j = 0; j < 8; ++j) {
      float mx = fmaxf(sv[0][j], sv[1][j]);
#pragma unroll
      for (int d = 8; d >= 1; d >>= 1) mx = fmaxf(mx, __shfl_xor(mx, d, 32));
      float mnew  = fmaxf(rmax[j], mx);
      float alpha = __expf(rmax[j] - mnew);
      float p0 = __expf(sv[0][j] - mnew);
      float p1 = __expf(sv[1][j] - mnew);
      float ps = p0 + p1;
#pragma unroll
      for (int d = 8; d >= 1; d >>= 1) ps += __shfl_xor(ps, d, 32);
      rsum[j] = rsum[j] * alpha + ps;
      rmax[j] = mnew;
#pragma unroll
      for (int t = 0; t < 4; ++t) o[t][j] *= alpha;
      int row = hi * 8 + j;                  // stage P via LDS (C->A re-layout)
      plds[row * 32 + cl]      = (_Float16)p0;
      plds[row * 32 + 16 + cl] = (_Float16)p1;
    }
    // read back P as 16x32 A-fragment (same wave: DS ops in-order, b128 loads)
    v16h pa;
    {
      const _Float16* pp = plds + (size_t)(lane & 15) * 32 + hi * 16;
      frag_cast t;
      t.u[0] = *(const uint4*)(pp);
      t.u[1] = *(const uint4*)(pp + 8);
      pa = t.h;
    }
    // ---- O += P . V : Vt rows are d, contiguous in n ----
#pragma unroll
    for (int t = 0; t < 4; ++t) {
      v16h vb = load_bT_frag(Vth + (size_t)(t * 16) * M_SZ + n0, M_SZ);
      o[t] = wmma16(pa, vb, o[t]);
    }
  }
  // normalize + store f16 head output into (B, M, H)
  _Float16* Oh = O + (size_t)b * M_SZ * H_SZ + (size_t)kh * D_SZ;
#pragma unroll
  for (int t = 0; t < 4; ++t)
#pragma unroll
    for (int j = 0; j < 8; ++j) {
      int m = m0 + hi * 8 + j;
      Oh[(size_t)m * H_SZ + t * 16 + cl] = (_Float16)(o[t][j] / rsum[j]);
    }
}

extern "C" void kernel_launch(void* const* d_in, const int* in_sizes, int n_in,
                              void* d_out, int out_size, void* d_ws, size_t ws_size,
                              hipStream_t stream) {
  const float* h     = (const float*)d_in[0];   // (4,1024,1024)
  const float* keype = (const float*)d_in[1];   // (1,64,1024)
  const float* Wq    = (const float*)d_in[2];   // (1024,1024)
  const float* Wk    = (const float*)d_in[3];
  const float* Wv    = (const float*)d_in[4];
  const float* Wspan = (const float*)d_in[5];   // (1024,32)
  const float* Wo    = (const float*)d_in[6];
  float* out = (float*)d_out;

  const int BM  = B_SZ * M_SZ;                  // 4096
  const int NH  = H_SZ * H_SZ;                  // 1048576
  const int NHH = BM * H_SZ;                    // 4194304

  // workspace carve-up
  char* w = (char*)d_ws;
  _Float16* hh   = (_Float16*)w;  w += (size_t)NHH * 2;            // h in f16
  _Float16* WqP  = (_Float16*)w;  w += (size_t)NH * 2;             // packed B
  _Float16* WkP  = (_Float16*)w;  w += (size_t)NH * 2;
  _Float16* WvP  = (_Float16*)w;  w += (size_t)NH * 2;
  _Float16* WoP  = (_Float16*)w;  w += (size_t)NH * 2;
  _Float16* WsP  = (_Float16*)w;  w += (size_t)(H_SZ * 2 * KH) * 2;
  _Float16* PeP  = (_Float16*)w;  w += (size_t)(D_SZ * M_SZ) * 2;
  _Float16* Q16  = (_Float16*)w;  w += (size_t)NHH * 2;            // (B,M,H)
  _Float16* K16  = (_Float16*)w;  w += (size_t)NHH * 2;            // (B,M,H)
  _Float16* Vt16 = (_Float16*)w;  w += (size_t)NHH * 2;            // (B,H,M)
  _Float16* Ob16 = (_Float16*)w;  w += (size_t)NHH * 2;            // (B,M,H)
  float*    SpanF= (float*)w;     w += (size_t)BM * 2 * KH * 4;    // (B,M,2K)

  // h -> f16
  cvt_f32_f16<<<dim3(NHH / 256), dim3(256), 0, stream>>>(h, hh, NHH);

  // pack weights + key_pe into WMMA-B tile layout (fused f32->f16)
  auto pack = [&](const float* src, _Float16* dst, int Kd, int N) {
    int thr = (Kd / 32) * (N / 16) * 32;
    pack_b_f16<<<dim3((thr + 255) / 256), dim3(256), 0, stream>>>(src, dst, Kd, N);
  };
  pack(Wq,    WqP, H_SZ, H_SZ);
  pack(Wk,    WkP, H_SZ, H_SZ);
  pack(Wv,    WvP, H_SZ, H_SZ);
  pack(Wo,    WoP, H_SZ, H_SZ);
  pack(Wspan, WsP, H_SZ, 2 * KH);
  pack(keype, PeP, D_SZ, M_SZ);    // key_pe is (D x M): B[k][n] layout directly

  // projections: waves = (BM/32) * (H/64) = 2048 -> 256 blocks of 8 waves
  {
    dim3 g(((BM / 32) * (H_SZ / 64)) / 8), blk(256);
    gemm_f16_wmma<4><<<g, blk, 0, stream>>>(hh, WqP, nullptr, Q16, nullptr, BM, H_SZ, H_SZ);
    gemm_f16_wmma<4><<<g, blk, 0, stream>>>(hh, WkP, nullptr, K16, nullptr, BM, H_SZ, H_SZ);
    gemm_f16_wmma<4><<<g, blk, 0, stream>>>(hh, WvP, nullptr, nullptr, Vt16, BM, H_SZ, H_SZ);
  }
  // span projection: N=32, NT=2 -> one strip; waves = BM/32 = 128 -> 16 blocks
  gemm_f16_wmma<2><<<dim3(16), dim3(256), 0, stream>>>(
      hh, WsP, SpanF, nullptr, nullptr, BM, 2 * KH, H_SZ);

  // fused attention: B*K*(M/16) = 4096 waves -> 512 blocks of 8 waves
  attn_fused<<<dim3(512), dim3(256), 0, stream>>>(Q16, K16, Vt16, SpanF, PeP, Ob16);

  // output projection -> f32 d_out
  gemm_f16_wmma<4><<<dim3(((BM / 32) * (H_SZ / 64)) / 8), dim3(256), 0, stream>>>(
      Ob16, WoP, out, nullptr, nullptr, BM, H_SZ, H_SZ);

  (void)in_sizes; (void)n_in; (void)out_size; (void)ws_size;
}
